// MultiScaleAttention_28295244546799
// MI455X (gfx1250) — compile-verified, matrix operand order checked by reading, not run on hardware
//
#include <hip/hip_runtime.h>
#include <hip/hip_bf16.h>

typedef __attribute__((ext_vector_type(16))) _Float16 v16h;
typedef __attribute__((ext_vector_type(8)))  float    v8f;
typedef __attribute__((ext_vector_type(4)))  unsigned int v4u;
typedef __attribute__((ext_vector_type(4)))  int v4i;

#define WS7   7
#define PP    49
#define BATCH 16
#define CDIM  384
#define NPIX  3136
#define MROWS (BATCH * NPIX)      // 50176
#define QKVN  1152
#define LHEAD 4
#define HDIM  32

#if defined(__HIP_DEVICE_COMPILE__) && \
    __has_builtin(__builtin_amdgcn_global_load_async_to_lds_b128) && \
    __has_builtin(__builtin_amdgcn_s_wait_asynccnt)
#define USE_ASYNC_LDS 1
// Typed casts: v4i pointee in global (AS1) / LDS (AS3) address spaces.
#define ASG(p) ((__attribute__((address_space(1))) v4i*)(void*)(p))
#define ASL(p) ((__attribute__((address_space(3))) v4i*)(void*)(p))
#else
#define USE_ASYNC_LDS 0
#endif

__device__ __forceinline__ v8f zero8() {
    v8f z;
#pragma unroll
    for (int i = 0; i < 8; ++i) z[i] = 0.0f;
    return z;
}

// Build a 16x32 f16 WMMA operand fragment from a row-major f16 LDS tile.
// CDNA5 16-bit A layout: lanes 0-15 hold K=0..7 (v0..3) / K=16..23 (v4..7);
// lanes 16-31 hold K=8..15 / K=24..31 -> two contiguous ds_load_b128 runs.
__device__ __forceinline__ v16h load_frag(const _Float16* T, int rs, int row0, int kb) {
    int lane = threadIdx.x & 31;
    int row  = row0 + (lane & 15);
    int koff = kb + ((lane & 16) ? 8 : 0);
    const _Float16* p = T + row * rs + koff;
    union { v4u u[2]; v16h h; } f;
    f.u[0] = *(const v4u*)(p);
    f.u[1] = *(const v4u*)(p + 16);
    return f.h;
}

// ---------------------------------------------------------------------------
// Prepass conversions
// ---------------------------------------------------------------------------
__global__ void cvt_f32_f16(const float* __restrict__ src, _Float16* __restrict__ dst, long n) {
    long t = (long)blockIdx.x * blockDim.x + threadIdx.x;
    if (t < n) dst[t] = (_Float16)src[t];
}

// B[K][N] (f32) -> Bt[N][K] (f16)
__global__ void transpose_f32_f16(const float* __restrict__ B, _Float16* __restrict__ Bt,
                                  int K, int N) {
    long t = (long)blockIdx.x * blockDim.x + threadIdx.x;
    if (t >= (long)K * N) return;
    int n = (int)(t % N), k = (int)(t / N);
    Bt[(size_t)n * K + k] = (_Float16)B[(size_t)k * N + n];
}

// ---------------------------------------------------------------------------
// Tiled WMMA GEMM on f16 operands: C[M,N] = A[M,K] * Bt[N,K]^T (+ bias)
// Tile 128x128x32, 256 threads (8 waves, each 32x64 -> 8 WMMA / K-step).
// M%128==0, N%128==0, K%32==0 for all uses here.
// ---------------------------------------------------------------------------
template <bool OUT_F16>
__global__ __launch_bounds__(256) void gemm_h(
    const _Float16* __restrict__ A,   // [M][K]
    const _Float16* __restrict__ Bt,  // [N][K]
    const float* __restrict__ bias,   // [N] or null
    void* __restrict__ Cout,
    int M, int N, int K) {
    __shared__ __align__(16) _Float16 As[128 * 32];
    __shared__ __align__(16) _Float16 Bs[128 * 32];

    const int tid  = threadIdx.x;
    const int lane = tid & 31;
    const int wave = tid >> 5;
    const int wm   = wave >> 1;          // 0..3  (rows)
    const int wn   = wave & 1;           // 0..1  (cols)
    const int m0   = blockIdx.x * 128;
    const int n0   = blockIdx.y * 128;

    v8f acc[2][4];
#pragma unroll
    for (int i = 0; i < 2; ++i)
#pragma unroll
        for (int j = 0; j < 4; ++j) acc[i][j] = zero8();

    for (int k0 = 0; k0 < K; k0 += 32) {
        // Stage 128x32 f16 tiles of A and Bt: 512 16-byte chunks each,
        // 2 chunks per thread per matrix.
#pragma unroll
        for (int j = 0; j < 2; ++j) {
            int c   = tid + j * 256;
            int row = c >> 2;
            int off = (c & 3) * 8;
            const _Float16* ga = A  + (size_t)(m0 + row) * K + k0 + off;
            const _Float16* gb = Bt + (size_t)(n0 + row) * K + k0 + off;
#if USE_ASYNC_LDS
            __builtin_amdgcn_global_load_async_to_lds_b128(ASG(ga), ASL(&As[row * 32 + off]), 0, 0);
            __builtin_amdgcn_global_load_async_to_lds_b128(ASG(gb), ASL(&Bs[row * 32 + off]), 0, 0);
#else
            *(v4u*)&As[row * 32 + off] = *(const v4u*)ga;
            *(v4u*)&Bs[row * 32 + off] = *(const v4u*)gb;
#endif
        }
#if USE_ASYNC_LDS
        __builtin_amdgcn_s_wait_asynccnt(0);
#endif
        __syncthreads();

        v16h a0 = load_frag(As, 32, wm * 32, 0);
        v16h a1 = load_frag(As, 32, wm * 32 + 16, 0);
#pragma unroll
        for (int j = 0; j < 4; ++j) {
            v16h b = load_frag(Bs, 32, wn * 64 + j * 16, 0);
            acc[0][j] = __builtin_amdgcn_wmma_f32_16x16x32_f16(
                false, a0, false, b, (short)0, acc[0][j], false, false);
            acc[1][j] = __builtin_amdgcn_wmma_f32_16x16x32_f16(
                false, a1, false, b, (short)0, acc[1][j], false, false);
        }
        __syncthreads();
    }

    // C/D layout: lane n = l&15, rows m = r + 8*(l>>4)
    const int nl   = lane & 15;
    const int moff = (lane >> 4) * 8;
#pragma unroll
    for (int i = 0; i < 2; ++i) {
#pragma unroll
        for (int j = 0; j < 4; ++j) {
            int n = n0 + wn * 64 + j * 16 + nl;
            float bv = (!OUT_F16 && bias) ? bias[n] : 0.0f;
#pragma unroll
            for (int r = 0; r < 8; ++r) {
                int m = m0 + wm * 32 + i * 16 + moff + r;
                if (OUT_F16) {
                    ((_Float16*)Cout)[(size_t)m * N + n] = (_Float16)acc[i][j][r];
                } else {
                    ((float*)Cout)[(size_t)m * N + n] = acc[i][j][r] + bv;
                }
            }
        }
    }
}

// ---------------------------------------------------------------------------
// Mean+max pooling over 2^i x 2^i blocks (f16 in/out, f32 math).
// pooled layout: [s(3)][b][oy][ox][c(128)]
// ---------------------------------------------------------------------------
__global__ void pool_kernel(const _Float16* __restrict__ temp, _Float16* __restrict__ pooled,
                            int scale_i, int oh, int ow) {
    const int kk = 1 << scale_i;
    const long total = (long)3 * BATCH * oh * ow * 128;
    for (long t = (long)blockIdx.x * blockDim.x + threadIdx.x; t < total;
         t += (long)gridDim.x * blockDim.x) {
        int c = (int)(t % 128); long u = t / 128;
        int ox = (int)(u % ow); u /= ow;
        int oy = (int)(u % oh); u /= oh;
        int b  = (int)(u % BATCH);
        int s  = (int)(u / BATCH);
        float sum = 0.0f, mx = -3.4e38f;
        for (int dy = 0; dy < kk; ++dy)
            for (int dx = 0; dx < kk; ++dx) {
                float v = (float)temp[((size_t)b * NPIX + (size_t)(oy * kk + dy) * 56 + (ox * kk + dx)) * QKVN
                                      + s * CDIM + scale_i * 128 + c];
                sum += v; mx = fmaxf(mx, v);
            }
        pooled[(((size_t)s * BATCH + b) * oh * ow + (size_t)oy * ow + ox) * 128 + c] =
            (_Float16)(sum / (float)(kk * kk) + mx);
    }
}

// ---------------------------------------------------------------------------
// Windowed attention, one wave32 block per (window, head).
//   S = Q Kt ; P = softmax(S*scale + bias) ; Y = P V
// 49x32 operands padded to 64; matmuls on v_wmma_f32_16x16x32_f16.
// ---------------------------------------------------------------------------
__global__ __launch_bounds__(32) void attn_kernel(
    const _Float16* __restrict__ src,     // temp_h (scale 0) or pooled (scale>0)
    const float* __restrict__ bias_table, // [169][4]
    _Float16* __restrict__ dst16,         // fused_h (scale 0)
    float* __restrict__ dst32,            // ybuf (scale>0)
    int scale_i, int oh, int ow) {
    __shared__ __align__(16) _Float16 qS[64 * 32];
    __shared__ __align__(16) _Float16 kS[64 * 32];
    __shared__ __align__(16) _Float16 vtS[32 * 64];   // transposed: [d][p]
    __shared__ __align__(16) float    sS[64 * 64];
    __shared__ __align__(16) _Float16 pS[64 * 64];

    const int lane = threadIdx.x;
    const int h    = blockIdx.y;
    const int nrw  = ow / WS7;
    const int nreg = (oh / WS7) * nrw;
    const int win  = blockIdx.x;
    const int b    = win / nreg;
    const int reg  = win % nreg;
    const int rh   = reg / nrw;
    const int rw   = reg % nrw;
    const float qscale = 0.17677669529663687f;  // 1/sqrt(32)
    const size_t sstride = (size_t)BATCH * oh * ow * 128;

    // Stage Q,K in 16-byte chunks (pure f16 copies), padded rows zero.
    for (int e = lane; e < 64 * 4; e += 32) {
        int p = e >> 2, c8 = (e & 3) * 8;
        v4u qv = {0, 0, 0, 0}, kv = {0, 0, 0, 0};
        if (p < PP) {
            int py = p / WS7, px = p % WS7;
            int oy = rh * WS7 + py, ox = rw * WS7 + px;
            if (scale_i == 0) {
                const _Float16* pr = src + ((size_t)b * NPIX + (size_t)oy * 56 + ox) * QKVN + h * HDIM;
                qv = *(const v4u*)(pr + c8);
                kv = *(const v4u*)(pr + CDIM + c8);
            } else {
                size_t idx = (((size_t)b * oh + oy) * ow + ox) * 128 + h * HDIM;
                qv = *(const v4u*)(src + idx + c8);
                kv = *(const v4u*)(src + idx + sstride + c8);
            }
        }
        *(v4u*)&qS[p * 32 + c8] = qv;
        *(v4u*)&kS[p * 32 + c8] = kv;
    }
    // Stage V transposed ([d][p]) elementwise.
    for (int e = lane; e < 64 * 32; e += 32) {
        int p = e >> 5, d = e & 31;
        _Float16 vv = (_Float16)0.0f;
        if (p < PP) {
            int py = p / WS7, px = p % WS7;
            int oy = rh * WS7 + py, ox = rw * WS7 + px;
            if (scale_i == 0) {
                vv = src[((size_t)b * NPIX + (size_t)oy * 56 + ox) * QKVN + 2 * CDIM + h * HDIM + d];
            } else {
                vv = src[(((size_t)b * oh + oy) * ow + ox) * 128 + 2 * sstride + h * HDIM + d];
            }
        }
        vtS[d * 64 + p] = vv;
    }
    __syncthreads();

    // S = Q * Kt  (Kt column n == K row n -> B-fragment from row-major kS)
    const int nl   = lane & 15;
    const int moff = (lane >> 4) * 8;
#pragma unroll
    for (int mt = 0; mt < 4; ++mt) {
        v16h a = load_frag(qS, 32, mt * 16, 0);
#pragma unroll
        for (int nt = 0; nt < 4; ++nt) {
            v16h bf = load_frag(kS, 32, nt * 16, 0);
            v8f c = zero8();
            c = __builtin_amdgcn_wmma_f32_16x16x32_f16(false, a, false, bf, (short)0, c, false, false);
            int n = nt * 16 + nl;
#pragma unroll
            for (int r = 0; r < 8; ++r)
                sS[(mt * 16 + moff + r) * 64 + n] = c[r];
        }
    }
    __syncthreads();

    // Softmax rows (scale folded in here, + relative position bias).
#pragma unroll
    for (int rr = 0; rr < 2; ++rr) {
        int r = lane + rr * 32;
        if (r < PP) {
            int yp = r / WS7, xp = r % WS7;
            float mx = -3.4e38f;
            for (int j = 0; j < PP; ++j) {
                int yq = j / WS7, xq = j - yq * WS7;
                int ridx = (yp - yq + WS7 - 1) * (2 * WS7 - 1) + (xp - xq + WS7 - 1);
                float v = sS[r * 64 + j] * qscale + bias_table[ridx * LHEAD + h];
                sS[r * 64 + j] = v;
                mx = fmaxf(mx, v);
            }
            float sum = 0.0f;
            for (int j = 0; j < PP; ++j) {
                float e = __expf(sS[r * 64 + j] - mx);
                sS[r * 64 + j] = e;
                sum += e;
            }
            float inv = 1.0f / sum;
            for (int j = 0; j < PP; ++j) pS[r * 64 + j] = (_Float16)(sS[r * 64 + j] * inv);
            for (int j = PP; j < 64; ++j) pS[r * 64 + j] = (_Float16)0.0f;
        } else {
            for (int j = 0; j < 64; ++j) pS[r * 64 + j] = (_Float16)0.0f;
        }
    }
    __syncthreads();

    // Y = P * V  (K=64 -> two chained WMMAs per 16x16 tile)
#pragma unroll
    for (int mt = 0; mt < 4; ++mt) {
        v16h a0 = load_frag(pS, 64, mt * 16, 0);
        v16h a1 = load_frag(pS, 64, mt * 16, 32);
#pragma unroll
        for (int nt = 0; nt < 2; ++nt) {
            v16h b0 = load_frag(vtS, 64, nt * 16, 0);
            v16h b1 = load_frag(vtS, 64, nt * 16, 32);
            v8f c = zero8();
            c = __builtin_amdgcn_wmma_f32_16x16x32_f16(false, a0, false, b0, (short)0, c, false, false);
            c = __builtin_amdgcn_wmma_f32_16x16x32_f16(false, a1, false, b1, (short)0, c, false, false);
            int d = nt * 16 + nl;
#pragma unroll
            for (int r = 0; r < 8; ++r) {
                int p = mt * 16 + moff + r;
                if (p < PP) {
                    int py = p / WS7, px = p % WS7;
                    int oy = rh * WS7 + py, ox = rw * WS7 + px;
                    if (scale_i == 0) {
                        dst16[((size_t)b * NPIX + (size_t)oy * 56 + ox) * CDIM + h * HDIM + d] =
                            (_Float16)c[r];
                    } else {
                        dst32[((((size_t)b * LHEAD + h) * oh + oy) * ow + ox) * HDIM + d] = c[r];
                    }
                }
            }
        }
    }
}

// ---------------------------------------------------------------------------
// Bilinear upsample (half-pixel centers, edge clamp) into fused_h slice.
// ybuf layout: [b][h][oy][ox][d] (f32)
// ---------------------------------------------------------------------------
__global__ void upsample_kernel(const float* __restrict__ ybuf, _Float16* __restrict__ fused,
                                int scale_i, int oh, int ow) {
    const long total = (long)BATCH * LHEAD * HDIM * 56 * 56;
    long t = (long)blockIdx.x * blockDim.x + threadIdx.x;
    if (t >= total) return;
    int x = (int)(t % 56);
    int y = (int)((t / 56) % 56);
    int d = (int)((t / (56 * 56)) % HDIM);
    int h = (int)((t / (56 * 56 * HDIM)) % LHEAD);
    int b = (int)(t / ((long)56 * 56 * HDIM * LHEAD));

    float f  = (float)(56 / oh);
    float sy = (y + 0.5f) / f - 0.5f;
    float sx = (x + 0.5f) / f - 0.5f;
    int y0 = (int)floorf(sy), x0 = (int)floorf(sx);
    float wy = sy - (float)y0, wx = sx - (float)x0;
    int y0c = min(max(y0, 0), oh - 1), y1c = min(max(y0 + 1, 0), oh - 1);
    int x0c = min(max(x0, 0), ow - 1), x1c = min(max(x0 + 1, 0), ow - 1);

    const float* base = ybuf + (((size_t)b * LHEAD + h) * oh) * (size_t)ow * HDIM;
    float v00 = base[((size_t)y0c * ow + x0c) * HDIM + d];
    float v01 = base[((size_t)y0c * ow + x1c) * HDIM + d];
    float v10 = base[((size_t)y1c * ow + x0c) * HDIM + d];
    float v11 = base[((size_t)y1c * ow + x1c) * HDIM + d];
    float v = (1.0f - wy) * ((1.0f - wx) * v00 + wx * v01)
            +         wy  * ((1.0f - wx) * v10 + wx * v11);

    fused[((size_t)b * NPIX + (size_t)y * 56 + x) * CDIM + (scale_i * LHEAD + h) * HDIM + d] =
        (_Float16)v;
}

// ---------------------------------------------------------------------------
extern "C" void kernel_launch(void* const* d_in, const int* in_sizes, int n_in,
                              void* d_out, int out_size, void* d_ws, size_t ws_size,
                              hipStream_t stream) {
    const float* x        = (const float*)d_in[0];
    const float* w_qkv    = (const float*)d_in[1];
    const float* w_proj   = (const float*)d_in[2];
    const float* b_proj   = (const float*)d_in[3];
    const float* rel_bias = (const float*)d_in[4];
    float* out = (float*)d_out;

    // f16 workspace region, then f32 region (all chunk sizes 16B-aligned).
    _Float16* wsh = (_Float16*)d_ws;
    size_t o = 0;
    _Float16* xh      = wsh + o; o += (size_t)MROWS * CDIM;        // x in f16
    _Float16* wTq     = wsh + o; o += (size_t)QKVN * CDIM;         // w_qkv^T f16 [n][k]
    _Float16* wTp     = wsh + o; o += (size_t)CDIM * CDIM;         // w_proj^T f16 [n][k]
    _Float16* temp_h  = wsh + o; o += (size_t)MROWS * QKVN;        // qkv projections (f16)
    _Float16* fused_h = wsh + o; o += (size_t)MROWS * CDIM;        // concat of 3 scales (f16)
    _Float16* pooled1 = wsh + o; o += (size_t)3 * BATCH * 28 * 28 * 128;
    _Float16* pooled2 = wsh + o; o += (size_t)3 * BATCH * 14 * 14 * 128;
    float* wsf = (float*)(wsh + o);
    float* ybuf1 = wsf;
    float* ybuf2 = wsf + (size_t)BATCH * LHEAD * 28 * 28 * HDIM;

    // 0. Prepass: x -> f16, weights -> transposed f16
    {
        long nx = (long)MROWS * CDIM;
        cvt_f32_f16<<<(int)((nx + 255) / 256), 256, 0, stream>>>(x, xh, nx);
        long nq = (long)CDIM * QKVN;
        transpose_f32_f16<<<(int)((nq + 255) / 256), 256, 0, stream>>>(w_qkv, wTq, CDIM, QKVN);
        long np = (long)CDIM * CDIM;
        transpose_f32_f16<<<(int)((np + 255) / 256), 256, 0, stream>>>(w_proj, wTp, CDIM, CDIM);
    }

    // 1. QKV projection GEMM (50176 x 1152 x 384) -> f16
    gemm_h<true><<<dim3(MROWS / 128, QKVN / 128), 256, 0, stream>>>(
        xh, wTq, nullptr, temp_h, MROWS, QKVN, CDIM);

    // 2. Pool scales 1 and 2 (mean + max)
    {
        long t1 = (long)3 * BATCH * 28 * 28 * 128;
        long t2 = (long)3 * BATCH * 14 * 14 * 128;
        pool_kernel<<<(int)((t1 + 255) / 256), 256, 0, stream>>>(temp_h, pooled1, 1, 28, 28);
        pool_kernel<<<(int)((t2 + 255) / 256), 256, 0, stream>>>(temp_h, pooled2, 2, 14, 14);
    }

    // 3. Windowed attention per scale (one wave per window-head)
    attn_kernel<<<dim3(BATCH * 64, LHEAD), 32, 0, stream>>>(temp_h,  rel_bias, fused_h, nullptr, 0, 56, 56);
    attn_kernel<<<dim3(BATCH * 16, LHEAD), 32, 0, stream>>>(pooled1, rel_bias, nullptr, ybuf1,  1, 28, 28);
    attn_kernel<<<dim3(BATCH * 4,  LHEAD), 32, 0, stream>>>(pooled2, rel_bias, nullptr, ybuf2,  2, 14, 14);

    // 4. Bilinear upsample scales 1,2 into fused channel slices
    {
        long tu = (long)BATCH * LHEAD * HDIM * 56 * 56;
        upsample_kernel<<<(int)((tu + 255) / 256), 256, 0, stream>>>(ybuf1, fused_h, 1, 28, 28);
        upsample_kernel<<<(int)((tu + 255) / 256), 256, 0, stream>>>(ybuf2, fused_h, 2, 14, 14);
    }

    // 5. Output projection GEMM (+bias) -> d_out (f32)
    gemm_h<false><<<dim3(MROWS / 128, CDIM / 128), 256, 0, stream>>>(
        fused_h, wTp, b_proj, out, MROWS, CDIM, CDIM);
}
